// Decoder_47751446397523
// MI455X (gfx1250) — matrix operand order repaired and checked by reading, not compiled, not run on hardware
//
#include <hip/hip_runtime.h>
#include <hip/hip_bf16.h>
#include <math.h>

// ---------------------------------------------------------------------------
// Problem constants (match reference)
// ---------------------------------------------------------------------------
#define BB 128
#define TT 32
#define NN 16
#define DD 1024
#define NEG_INF -100000.0f

typedef __attribute__((ext_vector_type(16))) _Float16 v16h;
typedef __attribute__((ext_vector_type(8)))  float    v8f;

__device__ __forceinline__ unsigned int pack2h(float a, float b) {
    union { _Float16 h[2]; unsigned int u; } p;
    p.h[0] = (_Float16)a;
    p.h[1] = (_Float16)b;
    return p.u;
}

// ---------------------------------------------------------------------------
// WMMA GEMM:  C[M,N] = A[M,K] @ W[K,N] (+bias[N]) (+C if ACC)
// fp32 in/out, f16 WMMA math (v_wmma_f32_16x16x32_f16, f32 accumulate).
//
// Block tile 128x128, 8 waves (256 thr), K-chunk 32 (one WMMA K-step).
// Wave (wrow 0..3, wcol 0..1) owns 32x64: 2 (M) x 4 (N) tiles = 8 WMMAs/chunk.
//
// Double-buffered LDS (ping-pong, 2x16KB) with global->register prefetch:
//   iter c: issue chunk c+1 global loads; WMMA on lds[c&1]; convert+store
//   regs to lds[(c+1)&1]; one barrier per iteration.
//
// LDS tiles are in *fragment layout*: each lane's 16-half fragment is one
// contiguous 32-byte run -> 2x ds_load_b128 per fragment.
//
// Fragment layouts (CDNA5 ISA 7.12.2, 16-bit, 16x16x32):
//  A 16x32:  lane l: row M=l%16, K-half=l/16; halfs e=2j+p:
//            K = ((j>>2)<<4) + (l/16)*8 + ((j&3)<<1) + p
//  B 32x16:  lane l: col N=l%16; halfs e=2j+p: K same as A
//  C 16x16 f32: VGPR v: M = v + (l/16)*8, N = l%16
//
// Requires M%128==0, N%128==0, K%32==0 (true for all calls here).
// ---------------------------------------------------------------------------
#define GBM 128
#define GBN 128
#define GKC 32

template <bool HAS_BIAS, bool ACC>
__global__ __launch_bounds__(256)
void gemm_f16_wmma(const float* __restrict__ A, const float* __restrict__ W,
                   const float* __restrict__ bias, float* __restrict__ C,
                   int M, int N, int K)
{
    __shared__ unsigned int sA32[2][2048];   // 2 x 8KB
    __shared__ unsigned int sB32[2][2048];   // 2 x 8KB

    const int tid  = threadIdx.x;
    const int lane = tid & 31;
    const int wid  = tid >> 5;
    const int wrow = wid >> 1;   // 0..3 -> 32-row band
    const int wcol = wid & 1;    // 0..1 -> 64-col band
    const int bm   = blockIdx.x * GBM;
    const int bn   = blockIdx.y * GBN;
    const int lm   = lane & 15;

    // ---- per-thread staging indices (constant across chunks) ----
    int aR[8], aKp[8], aLds[8];
    int bKp[8], bCol[8], bLds[8];
#pragma unroll
    for (int p = 0; p < 8; ++p) {
        int i  = tid + (p << 8);
        int r  = i >> 4, kp = i & 15;                 // A: 128 rows x 16 kpairs
        aR[p]  = r; aKp[p] = kp;
        aLds[p] = (r >> 4) * 256 + ((r & 15) + (((kp >> 2) & 1) << 4)) * 8
                + ((kp >> 3) << 2) + (kp & 3);
        int kq = i >> 7, col = i & 127;               // B: 16 kpairs x 128 cols
        bKp[p] = kq; bCol[p] = col;
        bLds[p] = (col >> 4) * 256 + ((col & 15) + (((kq >> 2) & 1) << 4)) * 8
                + ((kq >> 3) << 2) + (kq & 3);
    }

    float2 aReg[8];
    float  bReg0[8], bReg1[8];

    auto load_chunk = [&](int k0) {
#pragma unroll
        for (int p = 0; p < 8; ++p)
            aReg[p] = *(const float2*)(A + (size_t)(bm + aR[p]) * K + k0 + (aKp[p] << 1));
#pragma unroll
        for (int p = 0; p < 8; ++p) {
            int k = bKp[p] << 1;
            bReg0[p] = W[(size_t)(k0 + k)     * N + bn + bCol[p]];
            bReg1[p] = W[(size_t)(k0 + k + 1) * N + bn + bCol[p]];
        }
    };
    auto store_lds = [&](int buf) {
#pragma unroll
        for (int p = 0; p < 8; ++p)
            sA32[buf][aLds[p]] = pack2h(aReg[p].x, aReg[p].y);
#pragma unroll
        for (int p = 0; p < 8; ++p)
            sB32[buf][bLds[p]] = pack2h(bReg0[p], bReg1[p]);
    };

    v8f acc[2][4];
#pragma unroll
    for (int mt = 0; mt < 2; ++mt)
#pragma unroll
        for (int nt = 0; nt < 4; ++nt)
#pragma unroll
            for (int j = 0; j < 8; ++j) acc[mt][nt][j] = 0.0f;

    // ---- pipeline prologue ----
    load_chunk(0);
    store_lds(0);
    __syncthreads();

    const int nCh = K / GKC;
    for (int c = 0; c < nCh; ++c) {
        const int cur = c & 1;
        const bool more = (c + 1 < nCh);
        if (more) load_chunk((c + 1) * GKC);   // global loads in flight over WMMA

        v16h afrag[2];
#pragma unroll
        for (int mt = 0; mt < 2; ++mt)
            afrag[mt] = *(const v16h*)(&sA32[cur][(wrow * 2 + mt) * 256 + lane * 8]);
#pragma unroll
        for (int nt = 0; nt < 4; ++nt) {
            v16h bfrag = *(const v16h*)(&sB32[cur][(wcol * 4 + nt) * 256 + lane * 8]);
#pragma unroll
            for (int mt = 0; mt < 2; ++mt) {
                acc[mt][nt] = __builtin_amdgcn_wmma_f32_16x16x32_f16(
                    false, afrag[mt], false, bfrag, (short)0, acc[mt][nt],
                    false, false);
            }
        }

        if (more) store_lds(cur ^ 1);
        __syncthreads();
    }

    // ---- straight-line epilogue (templated: no runtime branches) ----
    const int rofs = (lane >> 4) * 8;
#pragma unroll
    for (int mt = 0; mt < 2; ++mt) {
#pragma unroll
        for (int nt = 0; nt < 4; ++nt) {
            const int col = bn + wcol * 64 + nt * 16 + lm;
            const int row0 = bm + wrow * 32 + mt * 16 + rofs;
            float* cp = C + (size_t)row0 * N + col;
            float bv = HAS_BIAS ? bias[col] : 0.0f;
            if (ACC) {
                float cv[8];
#pragma unroll
                for (int v = 0; v < 8; ++v) cv[v] = cp[(size_t)v * N];
#pragma unroll
                for (int v = 0; v < 8; ++v)
                    cp[(size_t)v * N] = acc[mt][nt][v] + bv + cv[v];
            } else {
#pragma unroll
                for (int v = 0; v < 8; ++v)
                    cp[(size_t)v * N] = acc[mt][nt][v] + bv;
            }
        }
    }
}

static inline void launch_gemm(const float* A, const float* W, const float* bias,
                               float* C, int M, int N, int K, bool acc,
                               hipStream_t s)
{
    dim3 grid(M / GBM, N / GBN);
    if (bias) {
        if (acc) gemm_f16_wmma<true,  true ><<<grid, 256, 0, s>>>(A, W, bias, C, M, N, K);
        else     gemm_f16_wmma<true,  false><<<grid, 256, 0, s>>>(A, W, bias, C, M, N, K);
    } else {
        if (acc) gemm_f16_wmma<false, true ><<<grid, 256, 0, s>>>(A, W, nullptr, C, M, N, K);
        else     gemm_f16_wmma<false, false><<<grid, 256, 0, s>>>(A, W, nullptr, C, M, N, K);
    }
}

// ---------------------------------------------------------------------------
// Attention scores: e[b,l] = sum_d tanh(Wh[b,d] + U[b,l,d]) * we[d] + be
//                            (+ -INF where mask[b,l]==0)
// ---------------------------------------------------------------------------
__global__ __launch_bounds__(256)
void attn_score_kernel(const float* __restrict__ Wh, const float* __restrict__ U,
                       const float* __restrict__ we, const float* __restrict__ be,
                       const unsigned char* __restrict__ mask,
                       float* __restrict__ e, int L, int D)
{
    __shared__ float red[256];
    const int bl = blockIdx.x;          // b*L + l
    const int b  = bl / L;
    const float* wh = Wh + (size_t)b * D;
    const float* u  = U  + (size_t)bl * D;
    float s = 0.0f;
    for (int d = threadIdx.x; d < D; d += 256)
        s += tanhf(wh[d] + u[d]) * we[d];
    red[threadIdx.x] = s;
    __syncthreads();
    for (int off = 128; off > 0; off >>= 1) {
        if (threadIdx.x < off) red[threadIdx.x] += red[threadIdx.x + off];
        __syncthreads();
    }
    if (threadIdx.x == 0) {
        float val = red[0] + be[0];
        if (mask && mask[bl] == 0) val += NEG_INF;
        e[bl] = val;
    }
}

// ---------------------------------------------------------------------------
// Softmax over L (<=32): one wave32 per batch row.
// ---------------------------------------------------------------------------
__global__ __launch_bounds__(32)
void softmax_kernel(const float* __restrict__ e, float* __restrict__ beta, int L)
{
    const int b = blockIdx.x;
    const int t = threadIdx.x;
    float v = (t < L) ? e[b * L + t] : -INFINITY;
    float m = v;
#pragma unroll
    for (int off = 16; off > 0; off >>= 1)
        m = fmaxf(m, __shfl_xor(m, off, 32));
    float ex = (t < L) ? __expf(v - m) : 0.0f;
    float s = ex;
#pragma unroll
    for (int off = 16; off > 0; off >>= 1)
        s += __shfl_xor(s, off, 32);
    if (t < L) beta[b * L + t] = ex / s;
}

// ---------------------------------------------------------------------------
// Generic weighted sum over axis L (float4-vectorized, F%4==0):
//   out[b*out_stride + out_off + 4f..] = sum_l X[(b*L+l)*F + 4f..] * w[b*L+l]
// Streams up to 512MB -> pure HBM bandwidth; prefetch upcoming frames.
// ---------------------------------------------------------------------------
__global__ __launch_bounds__(256)
void weighted_sum_kernel(const float* __restrict__ X, const float* __restrict__ w,
                         float* __restrict__ out, int L, int F,
                         int out_stride, int out_off, long total4)
{
    long idx = (long)blockIdx.x * blockDim.x + threadIdx.x;   // over B*F/4
    if (idx >= total4) return;
    int F4 = F >> 2;
    int b  = (int)(idx / F4);
    int f4 = (int)(idx % F4);
    const float4* xp = (const float4*)(X + ((size_t)b * L) * F) + f4;
    const float*  wp = w + (size_t)b * L;
    size_t strideF4 = (size_t)F4;
    float4 s; s.x = s.y = s.z = s.w = 0.0f;
    for (int l = 0; l < L; ++l) {
        if (l + 4 < L)
            __builtin_prefetch(xp + (size_t)(l + 4) * strideF4, 0, 0);
        float4 v = xp[(size_t)l * strideF4];
        float  c = wp[l];
        s.x += v.x * c; s.y += v.y * c; s.z += v.z * c; s.w += v.w * c;
    }
    float4* op = (float4*)(out + (size_t)b * out_stride + out_off) + f4;
    *op = s;
}

// ---------------------------------------------------------------------------
// LSTM pointwise update + masked state blend + 3 concatenated outputs:
// d_out = [h_new (B*D) | h_new (B*D) | c_new (B*D)] ; gates split i,f,g,o.
// ---------------------------------------------------------------------------
__device__ __forceinline__ float sigmoidf_(float x) {
    return 1.0f / (1.0f + __expf(-x));
}

__global__ __launch_bounds__(256)
void lstm_kernel(const float* __restrict__ gates, const float* __restrict__ c_in,
                 const float* __restrict__ h_in, const float* __restrict__ word_mask,
                 float* __restrict__ out, int B, int D)
{
    int idx = blockIdx.x * blockDim.x + threadIdx.x;   // over B*D
    if (idx >= B * D) return;
    int b = idx / D, d = idx % D;
    const float* g = gates + (size_t)b * 4 * D;
    float ig = g[d], fg = g[D + d], gg = g[2 * D + d], og = g[3 * D + d];
    float c  = c_in[idx];
    float cn = sigmoidf_(fg) * c + sigmoidf_(ig) * tanhf(gg);
    float hn = sigmoidf_(og) * tanhf(cn);
    float wm = word_mask[b];
    cn = wm * cn + (1.0f - wm) * c;
    hn = wm * hn + (1.0f - wm) * h_in[idx];
    size_t bd = (size_t)B * D;
    out[idx]          = hn;   // h_new
    out[bd + idx]     = hn;   // h_new[None]
    out[2 * bd + idx] = cn;   // c_new[None]
}

// ---------------------------------------------------------------------------
// Host-side launch sequence
// ---------------------------------------------------------------------------
extern "C" void kernel_launch(void* const* d_in, const int* in_sizes, int n_in,
                              void* d_out, int out_size, void* d_ws, size_t ws_size,
                              hipStream_t stream)
{
    (void)in_sizes; (void)n_in; (void)out_size; (void)ws_size;
    const int B = BB, T = TT, N = NN, D = DD;

    // ---- inputs (setup_inputs order) ----
    const float* res2d    = (const float*)d_in[0];   // [B,T,D]
    const float* i3d      = (const float*)d_in[1];   // [B,T,D]
    const float* rel_f    = (const float*)d_in[2];   // [B,T,N,D]
    const float* obj_f    = (const float*)d_in[3];   // [B,T,N,2D]
    const float* word     = (const float*)d_in[4];   // [B,D]
    const float* state_h  = (const float*)d_in[5];   // [1,B,D]
    const float* state_c  = (const float*)d_in[6];   // [1,B,D]
    const unsigned char* res_mask = (const unsigned char*)d_in[7];   // [B,T] bool
    const unsigned char* i3d_mask = (const unsigned char*)d_in[8];   // [B,T] bool
    const float* word_mask = (const float*)d_in[9];  // [B,1]
    const float* W_ha    = (const float*)d_in[10];
    const float* b_ha    = (const float*)d_in[11];
    const float* W_res_a = (const float*)d_in[12];
    const float* b_res_a = (const float*)d_in[13];
    const float* w_res_e = (const float*)d_in[14];
    const float* b_res_e = (const float*)d_in[15];
    const float* W_i3d_a = (const float*)d_in[16];
    const float* b_i3d_a = (const float*)d_in[17];
    const float* w_i3d_e = (const float*)d_in[18];
    const float* b_i3d_e = (const float*)d_in[19];
    const float* W_rel_a = (const float*)d_in[20];
    const float* b_rel_a = (const float*)d_in[21];
    const float* w_rel_e = (const float*)d_in[22];
    const float* b_rel_e = (const float*)d_in[23];
    const float* W_ls = (const float*)d_in[24];      // [2D,4D]
    const float* W_lr = (const float*)d_in[25];      // [3D,4D]
    const float* W_lw = (const float*)d_in[26];      // [D,4D]
    const float* W_lh = (const float*)d_in[27];      // [D,4D]
    const float* b_l  = (const float*)d_in[28];      // [4D]

    // ---- workspace layout (fp32 elements) ----
    float* ws = (float*)d_ws;
    size_t off = 0;
    float* Wh       = ws + off; off += (size_t)B * D;          // [B,D]
    float* Ures     = ws + off; off += (size_t)B * T * D;      // [B,T,D]
    float* Ui3d     = ws + off; off += (size_t)B * T * D;      // [B,T,D]
    float* e_res    = ws + off; off += (size_t)B * T;
    float* beta_res = ws + off; off += (size_t)B * T;
    float* e_i3d    = ws + off; off += (size_t)B * T;
    float* beta_i3d = ws + off; off += (size_t)B * T;
    float* scene    = ws + off; off += (size_t)B * 2 * D;      // [B,2D]
    float* rel      = ws + off; off += (size_t)B * N * D;      // [B,N,D]
    float* obj      = ws + off; off += (size_t)B * N * 2 * D;  // [B,N,2D]
    float* Urel     = ws + off; off += (size_t)B * N * D;      // [B,N,D]
    float* e_rel    = ws + off; off += (size_t)B * N;
    float* beta_rel = ws + off; off += (size_t)B * N;
    float* relf     = ws + off; off += (size_t)B * 3 * D;      // [B,3D]
    float* gates    = ws + off; off += (size_t)B * 4 * D;      // [B,4D]

    const float* h = state_h;   // [B,D]
    const float* c = state_c;   // [B,D]

    // 1) Wh = h @ W_ha + b_ha                       (M=128, N=1024, K=1024)
    launch_gemm(h, W_ha, b_ha, Wh, B, D, D, false, stream);

    // 2) Ures = res2d @ W_res_a + b_res_a           (M=4096, N=1024, K=1024)
    launch_gemm(res2d, W_res_a, b_res_a, Ures, B * T, D, D, false, stream);

    // 3) Ui3d = i3d @ W_i3d_a + b_i3d_a
    launch_gemm(i3d, W_i3d_a, b_i3d_a, Ui3d, B * T, D, D, false, stream);

    // 4) e_res / beta_res
    attn_score_kernel<<<B * T, 256, 0, stream>>>(Wh, Ures, w_res_e, b_res_e,
                                                 res_mask, e_res, T, D);
    softmax_kernel<<<B, 32, 0, stream>>>(e_res, beta_res, T);

    // 5) e_i3d / beta_i3d
    attn_score_kernel<<<B * T, 256, 0, stream>>>(Wh, Ui3d, w_i3d_e, b_i3d_e,
                                                 i3d_mask, e_i3d, T, D);
    softmax_kernel<<<B, 32, 0, stream>>>(e_i3d, beta_i3d, T);

    // 6) att_res -> scene[:, 0:D];  att_i3d -> scene[:, D:2D]
    {
        long tot4 = (long)B * D / 4;
        int blocks = (int)((tot4 + 255) / 256);
        weighted_sum_kernel<<<blocks, 256, 0, stream>>>(
            res2d, beta_res, scene, T, D, 2 * D, 0, tot4);
        weighted_sum_kernel<<<blocks, 256, 0, stream>>>(
            i3d, beta_i3d, scene, T, D, 2 * D, D, tot4);
    }

    // 7) rel[b,n,d] = sum_t relation_feats * beta_res   (F = N*D)
    {
        long tot4 = (long)B * N * D / 4;
        int blocks = (int)((tot4 + 255) / 256);
        weighted_sum_kernel<<<blocks, 256, 0, stream>>>(
            rel_f, beta_res, rel, T, N * D, N * D, 0, tot4);
    }
    // 8) obj[b,n,e] = sum_t object_feats * beta_res     (F = N*2D)
    {
        long tot4 = (long)B * N * 2 * D / 4;
        int blocks = (int)((tot4 + 255) / 256);
        weighted_sum_kernel<<<blocks, 256, 0, stream>>>(
            obj_f, beta_res, obj, T, N * 2 * D, N * 2 * D, 0, tot4);
    }

    // 9) Urel = rel @ W_rel_a + b_rel_a             (M=2048, N=1024, K=1024)
    launch_gemm(rel, W_rel_a, b_rel_a, Urel, B * N, D, D, false, stream);

    // 10) e_rel / beta_rel over N (no mask)
    attn_score_kernel<<<B * N, 256, 0, stream>>>(Wh, Urel, w_rel_e, b_rel_e,
                                                 nullptr, e_rel, N, D);
    softmax_kernel<<<B, 32, 0, stream>>>(e_rel, beta_rel, N);

    // 11) att_rel -> relf[:, 0:D];  att_obj -> relf[:, D:3D]
    {
        long tot4 = (long)B * D / 4;
        int blocks = (int)((tot4 + 255) / 256);
        weighted_sum_kernel<<<blocks, 256, 0, stream>>>(
            rel, beta_rel, relf, N, D, 3 * D, 0, tot4);
    }
    {
        long tot4 = (long)B * 2 * D / 4;
        int blocks = (int)((tot4 + 255) / 256);
        weighted_sum_kernel<<<blocks, 256, 0, stream>>>(
            obj, beta_rel, relf, N, 2 * D, 3 * D, D, tot4);
    }

    // 12) gates = scene@W_ls + relf@W_lr + word@W_lw + h@W_lh + b_l  (M=128, N=4096)
    launch_gemm(scene, W_ls, b_l,     gates, B, 4 * D, 2 * D, false, stream);
    launch_gemm(relf,  W_lr, nullptr, gates, B, 4 * D, 3 * D, true,  stream);
    launch_gemm(word,  W_lw, nullptr, gates, B, 4 * D, D,     true,  stream);
    launch_gemm(h,     W_lh, nullptr, gates, B, 4 * D, D,     true,  stream);

    // 13) LSTM update + masked blend + outputs
    lstm_kernel<<<(B * D + 255) / 256, 256, 0, stream>>>(
        gates, c, h, word_mask, (float*)d_out, B, D);
}
